// LSTM_AR_77549929496854
// MI455X (gfx1250) — compile-verified, matrix operand order checked by reading, not run on hardware
//
#include <hip/hip_runtime.h>
#include <hip/hip_bf16.h>

// ---------------------------------------------------------------------------
// 2-layer LSTM + autoregressive rollout, B=1024, H=256, T=256, WINDOW=128.
// MI455X / gfx1250 strategy:
//  * batch is the only parallel axis -> 64 workgroups x 16 batch rows, each
//    runs the full 256-step recurrence locally (no inter-WG sync).
//  * GEMMs via v_wmma_f32_16x16x32_bf16; weights pre-packed to bf16 WMMA
//    B-fragment order (2 aligned global_load_b128 per fragment, L2-resident).
//  * K=8 input projection folded into WMMA via zero-padded x fragment.
//  * cell state c in VGPRs across all steps; h tiles in LDS bf16 so
//    A-fragments are two aligned ds_load_b128 per lane.
//  * weight base addresses are laundered through asm each time step (stops
//    LICM hoisting ~200 B-fragments -> scratch spills, seen in round 1) and
//    cast to address_space(1) so loads are global_load_b128, not flat
//    (flat double-counts LOADcnt+DScnt; seen in round 2).
// ---------------------------------------------------------------------------

typedef __attribute__((ext_vector_type(16))) __bf16        v16bf;
typedef __attribute__((ext_vector_type(8)))  float         v8f;
typedef __attribute__((ext_vector_type(4)))  unsigned int  v4u;

typedef __attribute__((address_space(1))) const v4u*   gp_v4u;
typedef __attribute__((address_space(1))) const float* gp_f32;

union Frag {
    v16bf v;
    v4u   q[2];
    unsigned short u[16];
};

#define HID   256
#define TSEQ  256
#define WIN   128
#define KT_H  8      // 256 / 32 K-tiles

__device__ __forceinline__ unsigned short f2bf(float f) {
    unsigned u = __builtin_bit_cast(unsigned, f);
    u += 0x7FFFu + ((u >> 16) & 1u);          // round-to-nearest-even
    return (unsigned short)(u >> 16);
}
__device__ __forceinline__ float bf2f(unsigned short s) {
    unsigned u = ((unsigned)s) << 16;
    return __builtin_bit_cast(float, u);
}
__device__ __forceinline__ float sigm(float x) { return 1.0f / (1.0f + __expf(-x)); }

// ---------------------------------------------------------------------------
// Pack weight W[N=1024][Ksrc] (row-major fp32) into bf16 WMMA B fragments.
// Fragment element e = ((nt*KT + kt)*32 + lane)*16 + j, with
//   n = nt*16 + lane%16,  grp = lane/16,  v = j/2, half = j%2,
//   k = kt*32 + (v<4 ? 2v : 16+2(v-4)) + half + grp*8      (zero if k>=Ksrc)
// ---------------------------------------------------------------------------
__global__ void pack_w_kernel(const float* __restrict__ W,
                              unsigned short* __restrict__ outp,
                              int Ksrc, int KT) {
    int e = blockIdx.x * 256 + threadIdx.x;
    int total = 64 * KT * 512;
    if (e >= total) return;
    int j    = e & 15;
    int lane = (e >> 4) & 31;
    int kt   = (e >> 9) % KT;
    int nt   = (e >> 9) / KT;
    int n    = nt * 16 + (lane & 15);
    int grp  = lane >> 4;
    int v    = j >> 1, half = j & 1;
    int kb   = (v < 4) ? (2 * v) : (16 + 2 * (v - 4));
    int k    = kt * 32 + kb + half + grp * 8;
    float val = (k < Ksrc) ? W[(size_t)n * Ksrc + k] : 0.0f;
    outp[e] = f2bf(val);
}

__global__ void bias_sum_kernel(const float* __restrict__ bi0, const float* __restrict__ bh0,
                                const float* __restrict__ bi1, const float* __restrict__ bh1,
                                float* __restrict__ bs0, float* __restrict__ bs1) {
    int i = blockIdx.x * 256 + threadIdx.x;
    if (i < 1024) { bs0[i] = bi0[i] + bh0[i]; bs1[i] = bi1[i] + bh1[i]; }
}

// ---------------------------------------------------------------------------
// Main persistent kernel: 64 blocks x 256 threads (8 wave32 per WG).
// Wave w owns hidden tiles {2w, 2w+1} (32 hidden units) in both layers.
// ---------------------------------------------------------------------------
__global__ __launch_bounds__(256)
void lstm_ar_main(const float* __restrict__ x,
                  const unsigned short* __restrict__ pWih0g,
                  const unsigned short* __restrict__ pWhh0g,
                  const unsigned short* __restrict__ pWih1g,
                  const unsigned short* __restrict__ pWhh1g,
                  const float* __restrict__ bsum0g,
                  const float* __restrict__ bsum1g,
                  const float* __restrict__ wling,
                  const float* __restrict__ blin,
                  float* __restrict__ out)
{
    __shared__ __align__(16) unsigned short h0s[16 * HID];   // bf16 bits
    __shared__ __align__(16) unsigned short h1s[16 * HID];
    __shared__ float pred_lds[16];
    __shared__ float partial[32];

    const int tid  = threadIdx.x;
    const int lane = tid & 31;
    const int wave = tid >> 5;
    const int col  = lane & 15;        // N-col within tile / M-row for A frags
    const int grp  = lane >> 4;
    const int wgb  = blockIdx.x * 16;  // batch row base
    const int ht0  = wave * 2;         // first owned hidden tile

    for (int i = tid; i < 16 * HID; i += 256) { h0s[i] = 0; h1s[i] = 0; }
    if (tid < 16) pred_lds[tid] = 0.0f;
    __syncthreads();

    // persistent cell state (registers): 2 hidden tiles per layer
    v8f c0[2], c1[2];
    for (int tI = 0; tI < 2; ++tI)
        for (int j = 0; j < 8; ++j) { c0[tI][j] = 0.0f; c1[tI][j] = 0.0f; }

    #pragma clang loop unroll(disable)
    for (int t = 0; t < TSEQ; ++t) {
        // Launder loop-invariant base addresses (defeats LICM / spilling),
        // then use address_space(1) pointers so loads lower to
        // global_load_b128 (LOADcnt only) instead of flat_load_b128.
        unsigned long long aWih0 = (unsigned long long)pWih0g;
        unsigned long long aWhh0 = (unsigned long long)pWhh0g;
        unsigned long long aWih1 = (unsigned long long)pWih1g;
        unsigned long long aWhh1 = (unsigned long long)pWhh1g;
        unsigned long long aBs0  = (unsigned long long)bsum0g;
        unsigned long long aBs1  = (unsigned long long)bsum1g;
        unsigned long long aWl   = (unsigned long long)wling;
        asm volatile("" : "+s"(aWih0), "+s"(aWhh0), "+s"(aWih1), "+s"(aWhh1),
                          "+s"(aBs0), "+s"(aBs1), "+s"(aWl));

        // =================== layer 0: gates = x@Wih0^T + h0@Whh0^T + b ======
        v8f acc[2][4];
        {
            gp_f32 bs0 = (gp_f32)aBs0;
            #pragma unroll
            for (int tI = 0; tI < 2; ++tI)
                #pragma unroll
                for (int g = 0; g < 4; ++g) {
                    float bv = bs0[(g * 16 + ht0 + tI) * 16 + col];
                    #pragma unroll
                    for (int j = 0; j < 8; ++j) acc[tI][g][j] = bv;
                }
        }

        // x fragment: K 0..7 = x_t (or [pred, x_t[1:]]), rest zero-padded
        Frag Ax;
        #pragma unroll
        for (int j = 0; j < 16; ++j) Ax.u[j] = 0;
        if (grp == 0) {
            const float* xp = x + (((size_t)(wgb + col)) * TSEQ + t) * 8;
            float xv[8];
            #pragma unroll
            for (int j = 0; j < 8; ++j) xv[j] = xp[j];
            if (t >= WIN) xv[0] = pred_lds[col];      // AR feedback
            #pragma unroll
            for (int j = 0; j < 8; ++j) Ax.u[j] = f2bf(xv[j]);
        }
        #pragma unroll
        for (int tI = 0; tI < 2; ++tI)
            #pragma unroll
            for (int g = 0; g < 4; ++g) {
                int nt = g * 16 + ht0 + tI;
                Frag B;
                gp_v4u bp = (gp_v4u)(aWih0 + (((size_t)nt * 32 + lane) * 16) * 2);
                B.q[0] = bp[0]; B.q[1] = bp[1];
                acc[tI][g] = __builtin_amdgcn_wmma_f32_16x16x32_bf16(
                    false, Ax.v, false, B.v, (short)0, acc[tI][g], false, false);
            }

        #pragma clang loop unroll(disable)
        for (int kt = 0; kt < KT_H; ++kt) {
            Frag A;
            int hb = col * HID + kt * 32 + grp * 8;
            A.q[0] = *(const v4u*)&h0s[hb];
            A.q[1] = *(const v4u*)&h0s[hb + 16];
            #pragma unroll
            for (int tI = 0; tI < 2; ++tI)
                #pragma unroll
                for (int g = 0; g < 4; ++g) {
                    int nt = g * 16 + ht0 + tI;
                    Frag B;
                    gp_v4u bp = (gp_v4u)(aWhh0 + ((((size_t)nt * KT_H + kt) * 32 + lane) * 16) * 2);
                    B.q[0] = bp[0]; B.q[1] = bp[1];
                    acc[tI][g] = __builtin_amdgcn_wmma_f32_16x16x32_bf16(
                        false, A.v, false, B.v, (short)0, acc[tI][g], false, false);
                }
        }
        __syncthreads();   // all waves done reading old h0s

        // cell update layer 0 (PyTorch gate order i,f,g,o), write new h0
        #pragma unroll
        for (int tI = 0; tI < 2; ++tI) {
            int n = (ht0 + tI) * 16 + col;
            #pragma unroll
            for (int j = 0; j < 8; ++j) {
                float cc = sigm(acc[tI][1][j]) * c0[tI][j]
                         + sigm(acc[tI][0][j]) * tanhf(acc[tI][2][j]);
                c0[tI][j] = cc;
                float hh = sigm(acc[tI][3][j]) * tanhf(cc);
                h0s[(j + 8 * grp) * HID + n] = f2bf(hh);
            }
        }
        __syncthreads();   // new h0s visible

        // =================== layer 1: gates = h0@Wih1^T + h1@Whh1^T + b =====
        {
            gp_f32 bs1 = (gp_f32)aBs1;
            #pragma unroll
            for (int tI = 0; tI < 2; ++tI)
                #pragma unroll
                for (int g = 0; g < 4; ++g) {
                    float bv = bs1[(g * 16 + ht0 + tI) * 16 + col];
                    #pragma unroll
                    for (int j = 0; j < 8; ++j) acc[tI][g][j] = bv;
                }
        }

        #pragma clang loop unroll(disable)
        for (int kt = 0; kt < KT_H; ++kt) {
            Frag A;
            int hb = col * HID + kt * 32 + grp * 8;
            A.q[0] = *(const v4u*)&h0s[hb];
            A.q[1] = *(const v4u*)&h0s[hb + 16];
            #pragma unroll
            for (int tI = 0; tI < 2; ++tI)
                #pragma unroll
                for (int g = 0; g < 4; ++g) {
                    int nt = g * 16 + ht0 + tI;
                    Frag B;
                    gp_v4u bp = (gp_v4u)(aWih1 + ((((size_t)nt * KT_H + kt) * 32 + lane) * 16) * 2);
                    B.q[0] = bp[0]; B.q[1] = bp[1];
                    acc[tI][g] = __builtin_amdgcn_wmma_f32_16x16x32_bf16(
                        false, A.v, false, B.v, (short)0, acc[tI][g], false, false);
                }
        }
        #pragma clang loop unroll(disable)
        for (int kt = 0; kt < KT_H; ++kt) {
            Frag A;
            int hb = col * HID + kt * 32 + grp * 8;
            A.q[0] = *(const v4u*)&h1s[hb];
            A.q[1] = *(const v4u*)&h1s[hb + 16];
            #pragma unroll
            for (int tI = 0; tI < 2; ++tI)
                #pragma unroll
                for (int g = 0; g < 4; ++g) {
                    int nt = g * 16 + ht0 + tI;
                    Frag B;
                    gp_v4u bp = (gp_v4u)(aWhh1 + ((((size_t)nt * KT_H + kt) * 32 + lane) * 16) * 2);
                    B.q[0] = bp[0]; B.q[1] = bp[1];
                    acc[tI][g] = __builtin_amdgcn_wmma_f32_16x16x32_bf16(
                        false, A.v, false, B.v, (short)0, acc[tI][g], false, false);
                }
        }
        __syncthreads();   // done reading old h1s

        #pragma unroll
        for (int tI = 0; tI < 2; ++tI) {
            int n = (ht0 + tI) * 16 + col;
            #pragma unroll
            for (int j = 0; j < 8; ++j) {
                float cc = sigm(acc[tI][1][j]) * c1[tI][j]
                         + sigm(acc[tI][0][j]) * tanhf(acc[tI][2][j]);
                c1[tI][j] = cc;
                float hh = sigm(acc[tI][3][j]) * tanhf(cc);
                h1s[(j + 8 * grp) * HID + n] = f2bf(hh);
            }
        }
        __syncthreads();   // new h1s visible

        // =================== linear head + pred feedback ====================
        if (wave == 0) {
            gp_f32 wl = (gp_f32)aWl;
            float s = 0.0f;
            #pragma clang loop unroll(disable)
            for (int kk = 0; kk < 8; ++kk) {
                int base = grp * 128 + kk * 16;
                Frag hq;
                hq.q[0] = *(const v4u*)&h1s[col * HID + base];
                hq.q[1] = *(const v4u*)&h1s[col * HID + base + 8];
                #pragma unroll
                for (int j = 0; j < 16; ++j)
                    s += bf2f(hq.u[j]) * wl[base + j];
            }
            partial[lane] = s;
        }
        __syncthreads();
        if (tid < 16) {
            float p = partial[tid] + partial[tid + 16] + blin[0];
            pred_lds[tid] = p;
            if (t >= WIN - 1)
                out[(size_t)(wgb + tid) * (TSEQ - WIN + 1) + (t - (WIN - 1))] = p;
        }
        __syncthreads();   // pred_lds ready for next step's x fragment
    }
}

// ---------------------------------------------------------------------------
extern "C" void kernel_launch(void* const* d_in, const int* in_sizes, int n_in,
                              void* d_out, int out_size, void* d_ws, size_t ws_size,
                              hipStream_t stream) {
    const float* x    = (const float*)d_in[0];
    const float* Wih0 = (const float*)d_in[1];
    const float* Whh0 = (const float*)d_in[2];
    const float* bih0 = (const float*)d_in[3];
    const float* bhh0 = (const float*)d_in[4];
    const float* Wih1 = (const float*)d_in[5];
    const float* Whh1 = (const float*)d_in[6];
    const float* bih1 = (const float*)d_in[7];
    const float* bhh1 = (const float*)d_in[8];
    const float* Wlin = (const float*)d_in[9];
    const float* blin = (const float*)d_in[10];
    float* out = (float*)d_out;

    char* ws = (char*)d_ws;
    unsigned short* pWih0 = (unsigned short*)(ws);                      //  64 KB
    unsigned short* pWhh0 = (unsigned short*)(ws + 65536);              // 512 KB
    unsigned short* pWih1 = (unsigned short*)(ws + 65536 + 524288);     // 512 KB
    unsigned short* pWhh1 = (unsigned short*)(ws + 65536 + 2 * 524288); // 512 KB
    float* bs0 = (float*)(ws + 65536 + 3 * 524288);                     //   4 KB
    float* bs1 = bs0 + 1024;                                            //   4 KB

    pack_w_kernel<<<(64 * 1 * 512 + 255) / 256, 256, 0, stream>>>(Wih0, pWih0, 8, 1);
    pack_w_kernel<<<(64 * 8 * 512 + 255) / 256, 256, 0, stream>>>(Whh0, pWhh0, 256, KT_H);
    pack_w_kernel<<<(64 * 8 * 512 + 255) / 256, 256, 0, stream>>>(Wih1, pWih1, 256, KT_H);
    pack_w_kernel<<<(64 * 8 * 512 + 255) / 256, 256, 0, stream>>>(Whh1, pWhh1, 256, KT_H);
    bias_sum_kernel<<<4, 256, 0, stream>>>(bih0, bhh0, bih1, bhh1, bs0, bs1);

    lstm_ar_main<<<64, 256, 0, stream>>>(x, pWih0, pWhh0, pWih1, pWhh1,
                                         bs0, bs1, Wlin, blin, out);
}
